// RandomAttention_45337674777145
// MI455X (gfx1250) — matrix operand order repaired and checked by reading, not compiled, not run on hardware
//
#include <hip/hip_runtime.h>

typedef __attribute__((ext_vector_type(16))) __bf16 v16bf;
typedef __attribute__((ext_vector_type(8)))  float  v8f;
typedef __attribute__((ext_vector_type(4)))  float  f4;
typedef __attribute__((ext_vector_type(4)))  int    i4;

static constexpr int   kH = 12;
static constexpr int   kS = 2048;
static constexpr int   kD = 64;
static constexpr float kNegInf = -1000000000.0f;

// ---------------------------------------------------------------------------
// Kernel 1: convert v (fp32, [H][S][D]) into bf16 WMMA B-fragments.
// Fragment = one 32x16 (KxN) tile for v_wmma_f32_16x16x32_bf16:
//   lanes 0-15  : N = lane,      elements 0..15 = K 0..15
//   lanes 16-31 : N = lane-16,   elements 0..15 = K 16..31
// Fragment id f = (h * (S/32) + kchunk) * 4 + ntile ; 512 bf16 per fragment,
// stored lane-major so kernel 2 loads one contiguous v16bf per lane.
// ---------------------------------------------------------------------------
__global__ void vfrag_convert_kernel(const float* __restrict__ v,
                                     __bf16* __restrict__ vfrag) {
    int idx = blockIdx.x * 256 + threadIdx.x;
    if (idx >= kH * kS * kD) return;
    int h   = idx / (kS * kD);
    int rem = idx - h * (kS * kD);
    int k   = rem / kD;
    int d   = rem - k * kD;

    int kc   = k >> 5;        // 32-wide K chunk
    int kin  = k & 31;
    int nt   = d >> 4;        // 16-wide N tile
    int n    = d & 15;
    int lane = ((kin >> 4) << 4) | n;   // lanes 16-31 hold K 16..31
    int e    = kin & 15;

    size_t f = (size_t)(h * (kS / 32) + kc) * 4 + nt;
    vfrag[f * 512 + lane * 16 + e] = (__bf16)v[idx];
}

// ---------------------------------------------------------------------------
// Kernel 2: fused masked softmax + (attn @ v) for one (head, 16-row) strip.
// 256 threads = 8 wave32s per workgroup. LDS stages the whole 16x2048 strip.
// ---------------------------------------------------------------------------
__launch_bounds__(256)
__global__ void rand_attn_kernel(const float* __restrict__ scores,
                                 const int*   __restrict__ mask,
                                 const __bf16* __restrict__ vfrag,
                                 float* __restrict__ out,
                                 float* __restrict__ attn) {
    __shared__ float smem[16 * kS];       // 128 KB strip (reused for reduction)
    __shared__ float rmax_s[16];
    __shared__ float rsum_s[16];

    const int t  = threadIdx.x;
    const int qb = blockIdx.x;            // 0..127
    const int h  = blockIdx.y;            // 0..11
    const int q0 = qb * 16;

    // ---- Phase A: masked load scores -> LDS (coalesced b128) ----
    const size_t sbase = ((size_t)h * kS + q0) * kS;
    const size_t mbase = (size_t)q0 * kS;
    for (int i = 0; i < 32; ++i) {
        int r = i >> 1;
        int c = ((i & 1) << 10) + (t << 2);
        f4 s = *(const f4*)(scores + sbase + (size_t)r * kS + c);
        i4 m = *(const i4*)(mask + mbase + (size_t)r * kS + c);
        s.x = m.x ? kNegInf : s.x;
        s.y = m.y ? kNegInf : s.y;
        s.z = m.z ? kNegInf : s.z;
        s.w = m.w ? kNegInf : s.w;
        *(f4*)(smem + r * kS + c) = s;
    }
    __syncthreads();

    // ---- Phase B: per-row max and sum(exp). 16 lanes per row. ----
    {
        int r = t >> 4, j = t & 15;
        const float* row = smem + r * kS;
        float mx = kNegInf;
        for (int it = 0; it < 32; ++it) {
            f4 s = *(const f4*)(row + ((j + (it << 4)) << 2));
            mx = fmaxf(mx, fmaxf(fmaxf(s.x, s.y), fmaxf(s.z, s.w)));
        }
        for (int mlane = 8; mlane >= 1; mlane >>= 1)
            mx = fmaxf(mx, __shfl_xor(mx, mlane, 32));
        float sum = 0.0f;
        for (int it = 0; it < 32; ++it) {
            f4 s = *(const f4*)(row + ((j + (it << 4)) << 2));
            sum += __expf(s.x - mx) + __expf(s.y - mx) +
                   __expf(s.z - mx) + __expf(s.w - mx);
        }
        for (int mlane = 8; mlane >= 1; mlane >>= 1)
            sum += __shfl_xor(sum, mlane, 32);
        if (j == 0) { rmax_s[r] = mx; rsum_s[r] = sum; }
    }
    __syncthreads();

    // ---- Phase C1: p = exp(s-max)/sum ; overwrite LDS; stream attn out ----
    float* attn_h = attn + ((size_t)h * kS + q0) * kS;
    for (int i = 0; i < 32; ++i) {
        int r = i >> 1;
        int c = ((i & 1) << 10) + (t << 2);
        float mx  = rmax_s[r];
        float inv = 1.0f / rsum_s[r];
        f4 s = *(const f4*)(smem + r * kS + c);
        f4 p;
        p.x = __expf(s.x - mx) * inv;
        p.y = __expf(s.y - mx) * inv;
        p.z = __expf(s.z - mx) * inv;
        p.w = __expf(s.w - mx) * inv;
        *(f4*)(smem + r * kS + c)         = p;
        *(f4*)(attn_h + (size_t)r * kS + c) = p;
    }
    __syncthreads();

    // ---- Phase C2: each wave WMMAs its 256-wide K slice against v ----
    const int w    = t >> 5;
    const int lane = t & 31;
    const int r    = lane & 15;
    const int koff = (lane >> 4) << 4;    // lanes>=16 start at K+16 (contiguous k order, same as vfrag)

    v8f acc[4] = {};                       // 16x64 f32 partial (4 N-tiles)
    const v16bf* fragp = (const v16bf*)vfrag;
    const int fbase = h * (kS / 32) * 4;

    for (int kc = 0; kc < 8; ++kc) {
        const int kbase = (w << 8) + (kc << 5);
        const float* prow = smem + r * kS + kbase + koff;
        f4 p0 = *(const f4*)(prow);
        f4 p1 = *(const f4*)(prow + 4);
        f4 p2 = *(const f4*)(prow + 8);
        f4 p3 = *(const f4*)(prow + 12);
        v16bf a;
        a[0]  = (__bf16)p0.x; a[1]  = (__bf16)p0.y;
        a[2]  = (__bf16)p0.z; a[3]  = (__bf16)p0.w;
        a[4]  = (__bf16)p1.x; a[5]  = (__bf16)p1.y;
        a[6]  = (__bf16)p1.z; a[7]  = (__bf16)p1.w;
        a[8]  = (__bf16)p2.x; a[9]  = (__bf16)p2.y;
        a[10] = (__bf16)p2.z; a[11] = (__bf16)p2.w;
        a[12] = (__bf16)p3.x; a[13] = (__bf16)p3.y;
        a[14] = (__bf16)p3.z; a[15] = (__bf16)p3.w;

        const int f = fbase + (kbase >> 5) * 4;
#pragma unroll
        for (int nt = 0; nt < 4; ++nt) {
            v16bf b = fragp[(size_t)(f + nt) * 32 + lane];
            acc[nt] = __builtin_amdgcn_wmma_f32_16x16x32_bf16(
                false, a, false, b, (short)0, acc[nt], false, false);
        }
    }
    __syncthreads();

    // ---- Cross-wave reduction of 16x64 partials (reuse strip LDS) ----
    float* part = smem;
#pragma unroll
    for (int nt = 0; nt < 4; ++nt) {
#pragma unroll
        for (int j = 0; j < 8; ++j) {
            int row = (lane < 16) ? j : (j + 8);   // C/D layout: lanes>=16 -> M+8
            int col = (nt << 4) + (lane & 15);
            part[(w << 10) + row * kD + col] = acc[nt][j];
        }
    }
    __syncthreads();

    float* out_h = out + ((size_t)h * kS + q0) * kD;
    for (int p = 0; p < 4; ++p) {
        int o = (p << 8) + t;                       // 0..1023 = row*64+col
        float sum = 0.0f;
#pragma unroll
        for (int ww = 0; ww < 8; ++ww) sum += part[(ww << 10) + o];
        out_h[o] = sum;
    }
}

// ---------------------------------------------------------------------------
extern "C" void kernel_launch(void* const* d_in, const int* in_sizes, int n_in,
                              void* d_out, int out_size, void* d_ws, size_t ws_size,
                              hipStream_t stream) {
    const float* v      = (const float*)d_in[0];   // [1,12,2048,64] fp32
    const float* scores = (const float*)d_in[1];   // [1,12,2048,2048] fp32
    // d_in[2] = len_q (== 2048), compile-time constant here
    const int*   mask   = (const int*)d_in[3];     // [1,1,2048,2048] int32

    float* out  = (float*)d_out;                      // first tuple element
    float* attn = out + (size_t)kH * kS * kD;         // second tuple element
    __bf16* vfrag = (__bf16*)d_ws;                    // ~3 MB of bf16 fragments

    vfrag_convert_kernel<<<(kH * kS * kD + 255) / 256, 256, 0, stream>>>(v, vfrag);
    rand_attn_kernel<<<dim3(kS / 16, kH), 256, 0, stream>>>(scores, mask, vfrag,
                                                            out, attn);
}
